// ARMABlock_90950227460135
// MI455X (gfx1250) — matrix operand order repaired and checked by reading, not compiled, not run on hardware
//
#include <hip/hip_runtime.h>
#include <hip/hip_bf16.h>
#include <math.h>

// ---------------- problem constants ----------------
static constexpr int Bn = 2;
static constexpr int Tn = 2048;
static constexpr int Cn = 1024;
static constexpr int Hn = 16;
static constexpr int KW = 31;       // conv taps
static constexpr int HD = Cn / Hn;  // 64
static constexpr int Mrows = Bn * Tn;  // 4096

// ---------------- types ----------------
typedef __bf16 bf16_t;
typedef __attribute__((ext_vector_type(16))) __bf16 v16bf;
typedef __attribute__((ext_vector_type(8)))  float  v8f;
typedef __attribute__((ext_vector_type(4)))  unsigned int v4u;
typedef __attribute__((ext_vector_type(8)))  unsigned int v8u;

__device__ __forceinline__ bf16_t f2bf(float f) {
    unsigned u = __builtin_bit_cast(unsigned, f);
    unsigned r = u + 0x7FFFu + ((u >> 16) & 1u);   // round-to-nearest-even
    unsigned short h = (unsigned short)(r >> 16);
    return __builtin_bit_cast(bf16_t, h);
}

__device__ __forceinline__ v8f wmma_bf16(v16bf a, v16bf b, v8f c) {
    // D = A(16x32) * B(32x16) + C, f32 accumulate
    return __builtin_amdgcn_wmma_f32_16x16x32_bf16(
        false, a, false, b, (short)0, c, false, false);
}

// async copy: 16 bytes global -> LDS, tracked by ASYNCcnt (gfx1250)
__device__ __forceinline__ void async_b128(unsigned lds_off, const bf16_t* gptr) {
    asm volatile("global_load_async_to_lds_b128 %0, %1, off"
                 :: "v"(lds_off), "v"(gptr) : "memory");
}
__device__ __forceinline__ unsigned lds_off32(const void* p) {
    return (unsigned)(unsigned long long)p;  // flat LDS aperture: addr[31:0] = offset
}

// build a 16-element bf16 A-fragment from two 16B rows chunks
__device__ __forceinline__ v16bf load_afrag(const bf16_t* arow, int k0, int half) {
    v4u alo = *(const v4u*)(arow + k0 + half * 16);
    v4u ahi = *(const v4u*)(arow + k0 + half * 16 + 8);
    v8u av;
    av[0] = alo[0]; av[1] = alo[1]; av[2] = alo[2]; av[3] = alo[3];
    av[4] = ahi[0]; av[5] = ahi[1]; av[6] = ahi[2]; av[7] = ahi[3];
    return __builtin_bit_cast(v16bf, av);
}

// ---------------- kernel 1: depthwise conv + bf16 casts of x ----------------
__global__ __launch_bounds__(256)
void conv_cast_kernel(const float* __restrict__ x, const float* __restrict__ conv_w,
                      const float* __restrict__ conv_b,
                      float* __restrict__ xconv, bf16_t* __restrict__ xb,
                      bf16_t* __restrict__ vt) {
    int idx = blockIdx.x * 256 + threadIdx.x;
    if (idx >= Bn * Tn * Cn) return;
    int c = idx % Cn;
    int t = (idx / Cn) % Tn;
    int b = idx / (Cn * Tn);
    const int pad = (KW - 1) / 2;
    float acc = conv_b[c];
    #pragma unroll 4
    for (int k = 0; k < KW; ++k) {
        int tt = t + k - pad;
        if (tt >= 0 && tt < Tn)
            acc += x[((size_t)b * Tn + tt) * Cn + c] * conv_w[c * KW + k];
    }
    xconv[idx] = acc;
    float xv = x[idx];
    xb[idx] = f2bf(xv);
    int h = c >> 6, n = c & 63;  // V head layout: vt[b][h][n][t]
    vt[(((size_t)b * Hn + h) * HD + n) * Tn + t] = f2bf(xv);
}

// ---------------- kernel 2: cast weights to bf16 -----------------------------
__global__ __launch_bounds__(256)
void cast_w_kernel(const float* __restrict__ w1, const float* __restrict__ w2,
                   const float* __restrict__ w3,
                   const float* __restrict__ qw, const float* __restrict__ kw,
                   bf16_t* __restrict__ o1, bf16_t* __restrict__ o2,
                   bf16_t* __restrict__ o3,
                   bf16_t* __restrict__ oq, bf16_t* __restrict__ ok) {
    int i = blockIdx.x * 256 + threadIdx.x;
    if (i >= Cn * Cn) return;
    o1[i] = f2bf(w1[i]);
    o2[i] = f2bf(w2[i]);
    o3[i] = f2bf(w3[i]);
    if (i < 2 * Hn * Cn) {  // 32 x 1024 projection weights
        oq[i] = f2bf(qw[i]);
        ok[i] = f2bf(kw[i]);
    }
}

// ---------------- kernel 3: Q/K projections via WMMA -------------------------
// Skinny GEMM: (B*T x C) x (64 x C)^T, N = 64 = [q_w(32) ; kv_w(32)].
// One wave per 16-token tile; 4 WMMAs per 32-wide k-step.
__global__ __launch_bounds__(256)
void qk_wmma_kernel(const bf16_t* __restrict__ xb,
                    const bf16_t* __restrict__ qwb, const bf16_t* __restrict__ kwb,
                    const float* __restrict__ q_b, const float* __restrict__ kv_b,
                    float* __restrict__ Qraw,   // (B*T, 32)
                    float* __restrict__ Kc) {   // (B, H, 2, T)
    const int lane = threadIdx.x & 31;
    const int wave = threadIdx.x >> 5;
    const int lm = lane & 15;
    const int half = lane >> 4;
    const int m0 = (blockIdx.x * 8 + wave) * 16;
    const bf16_t* Arow = xb + (size_t)(m0 + lm) * Cn;
    v8f acc[4] = {{}, {}, {}, {}};
    #pragma unroll 2
    for (int k0 = 0; k0 < Cn; k0 += 32) {
        v16bf a = load_afrag(Arow, k0, half);
        #pragma unroll
        for (int nt = 0; nt < 4; ++nt) {
            const bf16_t* wsrc = (nt < 2) ? qwb : kwb;
            int nrow = (nt & 1) * 16 + lm;  // row within its 32-row weight matrix
            v16bf b = __builtin_bit_cast(
                v16bf, *(const v8u*)(wsrc + (size_t)nrow * Cn + k0 + half * 16));
            acc[nt] = wmma_bf16(a, b, acc[nt]);
        }
    }
    #pragma unroll
    for (int r = 0; r < 8; ++r) {
        int row = m0 + r + half * 8;          // token index in [0, B*T)
        int b = row / Tn, t = row % Tn;
        #pragma unroll
        for (int nt = 0; nt < 4; ++nt) {
            int j = (nt & 1) * 16 + lm;       // 0..31 within q or kv block
            float v = acc[nt][r];
            if (nt < 2) {
                Qraw[(size_t)row * 32 + j] = v + q_b[j];
            } else {
                float vv = v + kv_b[j];
                int h = j >> 1, d = j & 1;
                Kc[(((size_t)b * Hn + h) * 2 + d) * Tn + t] = vv;
            }
        }
    }
}

// ---------------- kernel 4: causal flash attention (P·V via WMMA) ------------
__global__ __launch_bounds__(128)
void attn_kernel(const float* __restrict__ Qraw,  // (B*T, 32)
                 const float* __restrict__ Kc,    // (B,H,2,T)
                 const bf16_t* __restrict__ Vt,   // (B,H,64,T)
                 float* __restrict__ Xatt) {      // (B,T,C)
    const int lane = threadIdx.x & 31;
    const int wave = threadIdx.x >> 5;
    const int lm = lane & 15;
    const int half = lane >> 4;
    const int tile = blockIdx.x * 4 + wave;     // over B*H*(T/16)
    const int tpb = Tn / 16;                    // 128 tiles per (b,h)
    const int bh = tile / tpb;
    const int t0 = (tile % tpb) * 16;
    const int b = bh / Hn, h = bh % Hn;

    const int trow = t0 + lm;                   // this lane's S-row
    const float q0 = Qraw[((size_t)b * Tn + trow) * 32 + 2 * h + 0];
    const float q1 = Qraw[((size_t)b * Tn + trow) * 32 + 2 * h + 1];
    const float* K0 = Kc + ((size_t)bh * 2 + 0) * Tn;
    const float* K1 = Kc + ((size_t)bh * 2 + 1) * Tn;
    const bf16_t* Vbh = Vt + (size_t)bh * HD * Tn;
    const float sc = rsqrtf((float)Tn);

    float mrun = -3.0e38f, lrun = 0.f;
    v8f acc0 = {}, acc1 = {}, acc2 = {}, acc3 = {};

    for (int s0 = 0; s0 <= t0 + 15; s0 += 32) {
        // scores; loads unconditional (s <= s0+31 <= T-1), mask via select
        float p[16];
        float mx = -3.0e38f;
        #pragma unroll
        for (int j = 0; j < 16; ++j) {
            int ko = half * 16 + j;
            int s = s0 + ko;
            float sv = (q0 * K0[s] + q1 * K1[s]) * sc;
            sv = (s <= trow) ? sv : -3.0e38f;
            p[j] = sv;
            mx = fmaxf(mx, sv);
        }
        mx = fmaxf(mx, __shfl_xor(mx, 16, 32));          // join row halves
        float mnew = fmaxf(mrun, mx);
        float corr = __expf(mrun - mnew);
        float lsum = 0.f;
        v16bf P;
        #pragma unroll
        for (int j = 0; j < 16; ++j) {
            float e = __expf(p[j] - mnew);
            lsum += e;
            P[j] = f2bf(e);
        }
        lsum += __shfl_xor(lsum, 16, 32);
        lrun = lrun * corr + lsum;
        mrun = mnew;
        // rescale accumulators per C-layout row (row = r + half*8)
        #pragma unroll
        for (int r = 0; r < 8; ++r) {
            float cfac = __shfl(corr, r + half * 8, 32);
            acc0[r] *= cfac; acc1[r] *= cfac; acc2[r] *= cfac; acc3[r] *= cfac;
        }
        // ---- P * V : 4 WMMAs (N = 64) ----
        const bf16_t* vbase = Vbh + (size_t)lm * Tn + s0 + half * 16;
        v16bf bv0 = __builtin_bit_cast(v16bf, *(const v8u*)(vbase + (size_t)0 * 16 * Tn));
        v16bf bv1 = __builtin_bit_cast(v16bf, *(const v8u*)(vbase + (size_t)1 * 16 * Tn));
        v16bf bv2 = __builtin_bit_cast(v16bf, *(const v8u*)(vbase + (size_t)2 * 16 * Tn));
        v16bf bv3 = __builtin_bit_cast(v16bf, *(const v8u*)(vbase + (size_t)3 * 16 * Tn));
        acc0 = wmma_bf16(P, bv0, acc0);
        acc1 = wmma_bf16(P, bv1, acc1);
        acc2 = wmma_bf16(P, bv2, acc2);
        acc3 = wmma_bf16(P, bv3, acc3);
    }
    // ---- normalize + store ----
    #pragma unroll
    for (int r = 0; r < 8; ++r) {
        int row = t0 + r + half * 8;
        float linv = 1.f / __shfl(lrun, r + half * 8, 32);
        size_t base = ((size_t)b * Tn + row) * Cn + h * HD + lm;
        Xatt[base + 0 * 16] = acc0[r] * linv;
        Xatt[base + 1 * 16] = acc1[r] * linv;
        Xatt[base + 2 * 16] = acc2[r] * linv;
        Xatt[base + 3 * 16] = acc3[r] * linv;
    }
}

// ---------------- WMMA GEMM with async-LDS double buffering ------------------
// D = A(MxK) * W(NxK)^T + bias, fused epilogue.
// Block = 256 thr (8 waves as 4M x 2N); block tile 64M x 128N; k-step 32.
// Stages copied global->LDS with GLOBAL_LOAD_ASYNC_TO_LDS_B128 (ASYNCcnt),
// double buffered; each wave computes 16Mx64N = 4 WMMAs per k-step.
// EPI 0: SiLU -> bf16 ; EPI 1: combine+decay -> bf16 ; EPI 2: bias -> f32
template <int EPI>
__global__ __launch_bounds__(256)
void gemm_bf16_kernel(const bf16_t* __restrict__ A, const bf16_t* __restrict__ W,
                      const float* __restrict__ bias,
                      const float* __restrict__ e0,  // x_conv   (EPI 1)
                      const float* __restrict__ e1,  // x_att    (EPI 1)
                      const float* __restrict__ e2,  // time_decay (EPI 1)
                      const float* __restrict__ e3,  // x        (EPI 1)
                      bf16_t* __restrict__ outb, float* __restrict__ outf) {
    __shared__ __align__(16) bf16_t As[2][64][32];    //  8 KB
    __shared__ __align__(16) bf16_t Bs[2][128][32];   // 16 KB
    const int tid  = threadIdx.x;
    const int lane = tid & 31;
    const int wave = tid >> 5;
    const int lm   = lane & 15;
    const int half = lane >> 4;
    const int wm   = wave & 3;   // 0..3  -> M subtile
    const int wn   = wave >> 2;  // 0..1  -> N subtile
    const int m0 = blockIdx.y * 64;
    const int n0 = blockIdx.x * 128;

    // cooperative copy assignments
    const int arow = tid >> 2, acol = (tid & 3) * 8;   // A: 64 rows x 4 x 16B
    const int brow = tid >> 1, bcol = (tid & 1) * 16;  // B: 128 rows x 2 x 32B
    const bf16_t* Ag = A + (size_t)(m0 + arow) * Cn + acol;
    const bf16_t* Bg = W + (size_t)(n0 + brow) * Cn + bcol;
    const unsigned aoff[2] = { lds_off32(&As[0][arow][acol]), lds_off32(&As[1][arow][acol]) };
    const unsigned boff[2] = { lds_off32(&Bs[0][brow][bcol]), lds_off32(&Bs[1][brow][bcol]) };

    v8f acc[4] = {{}, {}, {}, {}};
    constexpr int NK = Cn / 32;  // 32 k-steps

    // prologue: stage 0
    async_b128(aoff[0], Ag);
    async_b128(boff[0], Bg);
    async_b128(boff[0] + 16, Bg + 8);

    for (int kt = 0; kt < NK; ++kt) {
        const int buf = kt & 1;
        if (kt + 1 < NK) {
            const bf16_t* ag = Ag + (kt + 1) * 32;
            const bf16_t* bg = Bg + (kt + 1) * 32;
            async_b128(aoff[buf ^ 1], ag);
            async_b128(boff[buf ^ 1], bg);
            async_b128(boff[buf ^ 1] + 16, bg + 8);
            asm volatile("s_wait_asynccnt 0x3" ::: "memory");  // stage kt landed
        } else {
            asm volatile("s_wait_asynccnt 0x0" ::: "memory");
        }
        __syncthreads();

        // operand fragments from LDS
        v16bf afrag = load_afrag(&As[buf][wm * 16 + lm][0], 0, half);
        #pragma unroll
        for (int nt = 0; nt < 4; ++nt) {
            const bf16_t* bp = &Bs[buf][wn * 64 + nt * 16 + lm][half * 16];
            v16bf bfrag = __builtin_bit_cast(v16bf, *(const v8u*)bp);
            acc[nt] = wmma_bf16(afrag, bfrag, acc[nt]);
        }
        __syncthreads();  // done reading buf before it is re-filled at kt+2
    }

    // fused epilogue (C layout: vgpr r -> row r + half*8, col lm)
    #pragma unroll
    for (int r = 0; r < 8; ++r) {
        int row = m0 + wm * 16 + r + half * 8;
        #pragma unroll
        for (int nt = 0; nt < 4; ++nt) {
            int n = n0 + wn * 64 + nt * 16 + lm;
            float v = acc[nt][r] + bias[n];
            size_t idx = (size_t)row * Cn + n;
            if (EPI == 0) {
                float s = v / (1.f + __expf(-v));      // SiLU
                outb[idx] = f2bf(s);
            } else if (EPI == 1) {
                float comb = e0[idx] + 0.5f * e1[idx] + 0.5f * v;
                float td = e2[n];
                comb = comb * td + e3[idx] * (1.f - td);
                outb[idx] = f2bf(comb);
            } else {
                outf[idx] = v;
            }
        }
    }
}

// ---------------- kernel: LayerNorm + residual -------------------------------
__global__ __launch_bounds__(256)
void ln_kernel(const float* __restrict__ proj, const float* __restrict__ x,
               const float* __restrict__ g, const float* __restrict__ be,
               float* __restrict__ out) {
    const int row = blockIdx.x;  // over B*T
    const float* pr = proj + (size_t)row * Cn;
    float s = 0.f, ss = 0.f;
    for (int c = threadIdx.x; c < Cn; c += 256) {
        float v = pr[c];
        s += v;
        ss += v * v;
    }
    #pragma unroll
    for (int off = 16; off; off >>= 1) {
        s  += __shfl_xor(s, off, 32);
        ss += __shfl_xor(ss, off, 32);
    }
    __shared__ float sbuf[8], ssbuf[8];
    __shared__ float mu_s, rs_s;
    if ((threadIdx.x & 31) == 0) {
        sbuf[threadIdx.x >> 5] = s;
        ssbuf[threadIdx.x >> 5] = ss;
    }
    __syncthreads();
    if (threadIdx.x == 0) {
        float S = 0.f, SS = 0.f;
        #pragma unroll
        for (int w = 0; w < 8; ++w) { S += sbuf[w]; SS += ssbuf[w]; }
        float mu = S / (float)Cn;
        float var = SS / (float)Cn - mu * mu;
        mu_s = mu;
        rs_s = rsqrtf(var + 1e-5f);
    }
    __syncthreads();
    for (int c = threadIdx.x; c < Cn; c += 256) {
        out[(size_t)row * Cn + c] =
            (pr[c] - mu_s) * rs_s * g[c] + be[c] + x[(size_t)row * Cn + c];
    }
}

// ---------------- host launcher ----------------
extern "C" void kernel_launch(void* const* d_in, const int* in_sizes, int n_in,
                              void* d_out, int out_size, void* d_ws, size_t ws_size,
                              hipStream_t stream) {
    const float* x       = (const float*)d_in[0];
    const float* conv_w  = (const float*)d_in[1];
    const float* conv_b  = (const float*)d_in[2];
    const float* q_w     = (const float*)d_in[3];
    const float* q_b     = (const float*)d_in[4];
    const float* kv_w    = (const float*)d_in[5];
    const float* kv_b    = (const float*)d_in[6];
    const float* mem_w1  = (const float*)d_in[7];
    const float* mem_b1  = (const float*)d_in[8];
    const float* mem_w2  = (const float*)d_in[9];
    const float* mem_b2  = (const float*)d_in[10];
    const float* out_w   = (const float*)d_in[11];
    const float* out_b   = (const float*)d_in[12];
    const float* ln_g    = (const float*)d_in[13];
    const float* ln_b    = (const float*)d_in[14];
    const float* tdecay  = (const float*)d_in[15];

    char* ws = (char*)d_ws;
    size_t off = 0;
    auto alloc = [&](size_t bytes) -> void* {
        off = (off + 255) & ~(size_t)255;
        void* p = ws + off;
        off += bytes;
        return p;
    };

    // region 0: bf16 x (8 MiB) + bf16 V^T (8 MiB); reused later as proj (16 MiB)
    bf16_t* xb   = (bf16_t*)alloc((size_t)Mrows * Cn * 2);
    bf16_t* vt   = (bf16_t*)alloc((size_t)Bn * Hn * HD * Tn * 2);
    float*  proj = (float*)xb;  // safe: xb dead after GEMM1, vt dead after attn

    float*  xconv = (float*)alloc((size_t)Mrows * Cn * 4);
    float*  xatt  = (float*)alloc((size_t)Mrows * Cn * 4);
    float*  Qraw  = (float*)alloc((size_t)Mrows * 2 * Hn * 4);
    float*  Kc    = (float*)alloc((size_t)Bn * Hn * 2 * Tn * 4);
    bf16_t* w1b   = (bf16_t*)alloc((size_t)Cn * Cn * 2);
    bf16_t* w2b   = (bf16_t*)alloc((size_t)Cn * Cn * 2);
    bf16_t* w3b   = (bf16_t*)alloc((size_t)Cn * Cn * 2);
    bf16_t* qwb   = (bf16_t*)alloc((size_t)2 * Hn * Cn * 2);
    bf16_t* kwb   = (bf16_t*)alloc((size_t)2 * Hn * Cn * 2);
    bf16_t* y1b   = (bf16_t*)alloc((size_t)Mrows * Cn * 2);
    bf16_t* combb = (bf16_t*)alloc((size_t)Mrows * Cn * 2);
    (void)ws_size; (void)in_sizes; (void)n_in; (void)out_size;

    // 1) depthwise conv + x casts (xb, vt)
    conv_cast_kernel<<<(Bn * Tn * Cn + 255) / 256, 256, 0, stream>>>(
        x, conv_w, conv_b, xconv, xb, vt);
    // 2) weight casts (w1/w2/w3 + q/kv projection weights)
    cast_w_kernel<<<(Cn * Cn + 255) / 256, 256, 0, stream>>>(
        mem_w1, mem_w2, out_w, q_w, kv_w, w1b, w2b, w3b, qwb, kwb);
    // 3) Q/K projections via WMMA (N = 64 stacked)
    qk_wmma_kernel<<<Mrows / (16 * 8), 256, 0, stream>>>(
        xb, qwb, kwb, q_b, kv_b, Qraw, Kc);
    // 4) flash attention (WMMA P*V)
    attn_kernel<<<(Bn * Hn * (Tn / 16)) / 4, 128, 0, stream>>>(Qraw, Kc, vt, xatt);
    // 5) y1 = silu(x W1^T + b1)
    gemm_bf16_kernel<0><<<dim3(Cn / 128, Mrows / 64), 256, 0, stream>>>(
        xb, w1b, mem_b1, nullptr, nullptr, nullptr, nullptr, y1b, nullptr);
    // 6) x_mem = y1 W2^T + b2; comb = conv + 0.5 att + 0.5 mem; decay blend
    gemm_bf16_kernel<1><<<dim3(Cn / 128, Mrows / 64), 256, 0, stream>>>(
        y1b, w2b, mem_b2, xconv, xatt, tdecay, x, combb, nullptr);
    // 7) proj = comb W3^T + b3
    gemm_bf16_kernel<2><<<dim3(Cn / 128, Mrows / 64), 256, 0, stream>>>(
        combb, w3b, out_b, nullptr, nullptr, nullptr, nullptr, nullptr, proj);
    // 8) LayerNorm + residual
    ln_kernel<<<Mrows, 256, 0, stream>>>(proj, x, ln_g, ln_b, (float*)d_out);
}
